// HeteroGraph_26809185862283
// MI455X (gfx1250) — compile-verified, matrix operand order checked by reading, not run on hardware
//
#include <hip/hip_runtime.h>

#define HDIM 128
#define LDS_STRIDE 136   // halves; multiple of 8 -> 16B-aligned fragment chunks

typedef __attribute__((ext_vector_type(16))) _Float16 v16h;
typedef __attribute__((ext_vector_type(8)))  _Float16 v8h;
typedef __attribute__((ext_vector_type(8)))  float    v8f;

// gemm mode bits (compile-time template parameter)
#define GM_ACC  1   // accumulate into C (else overwrite)
#define GM_BIAS 2   // add bias[n]
#define GM_RELU 4   // apply relu at the end (only valid on the final pass for a buffer)

// ---------------- utility kernels ----------------

__global__ void zero_f32(float* __restrict__ p, long long n) {
  long long i = (long long)blockIdx.x * blockDim.x + threadIdx.x;
  if (i < n) p[i] = 0.f;
}

// out = a + b + c + d (elementwise), pre-sums the 4 W_root mats for operator dst
__global__ void sum4_f32(const float* __restrict__ a, const float* __restrict__ b,
                         const float* __restrict__ c, const float* __restrict__ d,
                         float* __restrict__ o, int n) {
  int i = blockIdx.x * blockDim.x + threadIdx.x;
  if (i < n) o[i] = a[i] + b[i] + c[i] + d[i];
}

// per-node-type input projection: out[n,c] = b[c] + sum_k x[n,k] * W[k,c]
__global__ void in_proj(const float* __restrict__ x, const float* __restrict__ W,
                        const float* __restrict__ b, float* __restrict__ out,
                        int nNodes, int K) {
  long long gid = (long long)blockIdx.x * blockDim.x + threadIdx.x;
  long long total = (long long)nNodes * HDIM;
  if (gid >= total) return;
  int c = (int)(gid & (HDIM - 1));
  long long node = gid >> 7;
  float acc = b[c];
  for (int k = 0; k < K; ++k) acc += x[node * K + k] * W[k * HDIM + c];
  out[gid] = acc;
}

// segment-sum via atomics: agg[dst[e], :] += h[src[e], :]
__global__ void edge_scatter(const float* __restrict__ h, const int* __restrict__ src,
                             const int* __restrict__ dst, float* __restrict__ agg, int nE) {
  long long gid = (long long)blockIdx.x * blockDim.x + threadIdx.x;
  long long total = (long long)nE * 32;
  if (gid >= total) return;
  int e  = (int)(gid >> 5);
  int c4 = (int)(gid & 31) * 4;
  int s = src[e], d = dst[e];
  const float4 v = *(const float4*)&h[(long long)s * HDIM + c4];
  float* p = &agg[(long long)d * HDIM + c4];
  atomicAdd(p + 0, v.x); atomicAdd(p + 1, v.y);
  atomicAdd(p + 2, v.z); atomicAdd(p + 3, v.w);
}

// pooled[batch[i], :] += h[i, :] ; cnt[batch[i]] += 1
__global__ void pool_scatter(const float* __restrict__ h, const int* __restrict__ batch,
                             float* __restrict__ pooled, float* __restrict__ cnt, int nNodes) {
  long long gid = (long long)blockIdx.x * blockDim.x + threadIdx.x;
  long long total = (long long)nNodes * 32;
  if (gid >= total) return;
  int i  = (int)(gid >> 5);
  int c4 = (int)(gid & 31) * 4;
  int b = batch[i];
  const float4 v = *(const float4*)&h[(long long)i * HDIM + c4];
  float* p = &pooled[(long long)b * HDIM + c4];
  atomicAdd(p + 0, v.x); atomicAdd(p + 1, v.y);
  atomicAdd(p + 2, v.z); atomicAdd(p + 3, v.w);
  if (c4 == 0) atomicAdd(&cnt[b], 1.0f);
}

// out[g] = bias + dot(pooled[g]/max(cnt[g],1), W)
__global__ void final_lin(const float* __restrict__ pooled, const float* __restrict__ cnt,
                          const float* __restrict__ W, const float* __restrict__ bias,
                          float* __restrict__ out, int nB) {
  int g = blockIdx.x * blockDim.x + threadIdx.x;
  if (g >= nB) return;
  float c = cnt[g]; c = c > 1.f ? c : 1.f;
  float inv = 1.f / c;
  float acc = bias[0];
  for (int k = 0; k < HDIM; ++k) acc += pooled[(long long)g * HDIM + k] * inv * W[k];
  out[g] = acc;
}

// ---------------- WMMA GEMM: C[N,128] (=|+=) A[N,128] @ W[128,128] (+bias)(relu) ----------
// f32 in memory, f16 in LDS, v_wmma_f32_16x16x32_f16 with f32 accumulate.
// MODE is compile-time so the epilogue specializes to straight-line code.
template <int MODE>
__global__ __launch_bounds__(256)
void gemm128(const float* __restrict__ A, const float* __restrict__ W,
             const float* __restrict__ bias, float* __restrict__ C, int nRows) {
  __shared__ __align__(16) _Float16 As[64 * LDS_STRIDE];     // A tile, row-major
  __shared__ __align__(16) _Float16 Bst[HDIM * LDS_STRIDE];  // W transposed: Bst[j][k] = W[k][j]

  const int tid  = threadIdx.x;
  const int row0 = blockIdx.x * 64;

  // cooperative load A tile (zero-pad past nRows), cvt f32->f16
  for (int i = tid; i < 64 * HDIM; i += 256) {
    int r = i >> 7, c = i & (HDIM - 1);
    float v = (row0 + r < nRows) ? A[(long long)(row0 + r) * HDIM + c] : 0.f;
    As[r * LDS_STRIDE + c] = (_Float16)v;
  }
  // cooperative load W transposed, cvt f32->f16
  for (int i = tid; i < HDIM * HDIM; i += 256) {
    int k = i >> 7, j = i & (HDIM - 1);
    Bst[j * LDS_STRIDE + k] = (_Float16)W[i];
  }
  __syncthreads();

  const int wave  = tid >> 5;          // 8 waves -> 8 n-tiles of 16 cols
  const int lane  = tid & 31;
  const int n0    = wave * 16;
  const int lhalf = lane >> 4;         // lane half selects k-subrange per WMMA layout
  const int lmod  = lane & 15;

  // B fragments for the 4 K-steps (constant across M-tiles)
  v16h bfrag[4];
#pragma unroll
  for (int kt = 0; kt < 4; ++kt) {
    int kbase = kt * 32 + lhalf * 8;
    v8h lo = *(const v8h*)&Bst[(n0 + lmod) * LDS_STRIDE + kbase];
    v8h hi = *(const v8h*)&Bst[(n0 + lmod) * LDS_STRIDE + kbase + 16];
    v16h b;
#pragma unroll
    for (int i = 0; i < 8; ++i) { b[i] = lo[i]; b[i + 8] = hi[i]; }
    bfrag[kt] = b;
  }

  const int n = n0 + lmod;
  const float bv = (MODE & GM_BIAS) ? bias[n] : 0.f;
  const bool fullTile = (row0 + 64 <= nRows);

#pragma unroll
  for (int mt = 0; mt < 4; ++mt) {
    v8f acc = {};
#pragma unroll
    for (int kt = 0; kt < 4; ++kt) {
      int kbase = kt * 32 + lhalf * 8;
      v8h lo = *(const v8h*)&As[(mt * 16 + lmod) * LDS_STRIDE + kbase];
      v8h hi = *(const v8h*)&As[(mt * 16 + lmod) * LDS_STRIDE + kbase + 16];
      v16h a;
#pragma unroll
      for (int i = 0; i < 8; ++i) { a[i] = lo[i]; a[i + 8] = hi[i]; }
      acc = __builtin_amdgcn_wmma_f32_16x16x32_f16(false, a, false, bfrag[kt],
                                                   (short)0, acc, false, false);
    }
    // D layout: lane -> n = n0+lmod ; rows m = mt*16 + lhalf*8 + j
    const long long rbase = (long long)(row0 + mt * 16 + lhalf * 8);
    if (fullTile) {
#pragma unroll
      for (int j = 0; j < 8; ++j) {
        long long idx = (rbase + j) * HDIM + n;
        float v = acc[j];
        if (MODE & GM_BIAS) v += bv;
        if (MODE & GM_ACC)  v += C[idx];
        if (MODE & GM_RELU) v = v > 0.f ? v : 0.f;
        C[idx] = v;
      }
    } else {
#pragma unroll
      for (int j = 0; j < 8; ++j) {
        if (rbase + j < nRows) {
          long long idx = (rbase + j) * HDIM + n;
          float v = acc[j];
          if (MODE & GM_BIAS) v += bv;
          if (MODE & GM_ACC)  v += C[idx];
          if (MODE & GM_RELU) v = v > 0.f ? v : 0.f;
          C[idx] = v;
        }
      }
    }
  }
}

// ---------------- host orchestration ----------------

static inline unsigned nblk(long long n, int t) { return (unsigned)((n + t - 1) / t); }

extern "C" void kernel_launch(void* const* d_in, const int* in_sizes, int n_in,
                              void* d_out, int out_size, void* d_ws, size_t ws_size,
                              hipStream_t stream) {
  (void)in_sizes; (void)n_in; (void)out_size; (void)ws_size;
  const int NB = 2048;
  // node types: 0=operator 1=table 2=column 3=predicate
  const int NT[4]   = {200000, 100000, 300000, 150000};
  // relations (RELS order): scannedby filters outputby connects calledby selfloop_table selfloop_column
  const int EN[7]   = {200000, 300000, 600000, 300000, 200000, 100000, 300000};
  const int SRCT[7] = {1, 3, 2, 2, 0, 1, 2};
  const int DSTT[7] = {0, 0, 0, 3, 0, 1, 2};
  // flat input indices (depth-first insertion order of setup_inputs dict)
  const int IX_X[4]    = {0, 1, 2, 3};
  const int IX_LINW[4] = {4, 6, 8, 10};
  const int IX_LINB[4] = {5, 7, 9, 11};
  const int XK[4]      = {4, 2, 10, 1};
  const int CONV[2]    = {12, 33};  // conv1/conv2 bases; rel r: +3r (W_rel), +3r+1 (b_rel), +3r+2 (W_root)
  const int IX_OUTW = 54, IX_OUTB = 55, EDGE0 = 56, IX_BATCH = 70;

  // workspace layout (floats)
  float* ws = (float*)d_ws;
  long long S[4];
  for (int t = 0; t < 4; ++t) S[t] = (long long)NT[t] * HDIM;
  float* hA[4]; float* hB[4];
  long long off = 0;
  for (int t = 0; t < 4; ++t) { hA[t] = ws + off; off += S[t]; }
  for (int t = 0; t < 4; ++t) { hB[t] = ws + off; off += S[t]; }
  float* agg    = ws + off; off += S[2];          // max dst count (column)
  float* wsum[2]; wsum[0] = ws + off; off += HDIM * HDIM;
  wsum[1] = ws + off; off += HDIM * HDIM;
  float* pooled = ws + off; off += (long long)NB * HDIM;
  float* cnt    = ws + off; off += NB;

  const int T = 256;

  // pre-sum the 4 operator-destination W_root matrices per layer
  for (int L = 0; L < 2; ++L) {
    const float* r0 = (const float*)d_in[CONV[L] + 3 * 0 + 2];  // scannedby
    const float* r1 = (const float*)d_in[CONV[L] + 3 * 1 + 2];  // filters
    const float* r2 = (const float*)d_in[CONV[L] + 3 * 2 + 2];  // outputby
    const float* r4 = (const float*)d_in[CONV[L] + 3 * 4 + 2];  // calledby
    sum4_f32<<<nblk(HDIM * HDIM, T), T, 0, stream>>>(r0, r1, r2, r4, wsum[L], HDIM * HDIM);
  }

  // input projections -> hA
  for (int t = 0; t < 4; ++t) {
    in_proj<<<nblk(S[t], T), T, 0, stream>>>((const float*)d_in[IX_X[t]],
                                             (const float*)d_in[IX_LINW[t]],
                                             (const float*)d_in[IX_LINB[t]],
                                             hA[t], NT[t], XK[t]);
  }

  for (int L = 0; L < 2; ++L) {
    float** cur = (L == 0) ? hA : hB;
    float** out = (L == 0) ? hB : hA;
    const int base = CONV[L];

    // root terms FIRST in overwrite mode (no out-buffer zeroing, no C read):
    // out_d = h_d @ W_root  (operator uses the pre-summed 4-matrix root)
    gemm128<0><<<nblk(NT[0], 64), T, 0, stream>>>(cur[0], wsum[L], (const float*)nullptr,
                                                  out[0], NT[0]);
    gemm128<0><<<nblk(NT[1], 64), T, 0, stream>>>(cur[1], (const float*)d_in[base + 3 * 5 + 2],
                                                  (const float*)nullptr, out[1], NT[1]);
    gemm128<0><<<nblk(NT[2], 64), T, 0, stream>>>(cur[2], (const float*)d_in[base + 3 * 6 + 2],
                                                  (const float*)nullptr, out[2], NT[2]);
    gemm128<0><<<nblk(NT[3], 64), T, 0, stream>>>(cur[3], (const float*)d_in[base + 3 * 3 + 2],
                                                  (const float*)nullptr, out[3], NT[3]);

    // relation terms accumulate; the last pass touching each dst type (r>=3 in this
    // order: connects->predicate, calledby->operator, selfloop_table, selfloop_column)
    // also applies the ReLU.
    for (int r = 0; r < 7; ++r) {
      const int st = SRCT[r], dt = DSTT[r];
      const long long aggN = (long long)NT[dt] * HDIM;
      zero_f32<<<nblk(aggN, T), T, 0, stream>>>(agg, aggN);
      edge_scatter<<<nblk((long long)EN[r] * 32, T), T, 0, stream>>>(
          cur[st], (const int*)d_in[EDGE0 + 2 * r], (const int*)d_in[EDGE0 + 2 * r + 1],
          agg, EN[r]);
      const float* Wrel = (const float*)d_in[base + 3 * r];
      const float* brel = (const float*)d_in[base + 3 * r + 1];
      if (r >= 3) {
        gemm128<GM_ACC | GM_BIAS | GM_RELU>
            <<<nblk(NT[dt], 64), T, 0, stream>>>(agg, Wrel, brel, out[dt], NT[dt]);
      } else {
        gemm128<GM_ACC | GM_BIAS>
            <<<nblk(NT[dt], 64), T, 0, stream>>>(agg, Wrel, brel, out[dt], NT[dt]);
      }
    }
  }

  // global mean pool over operator nodes (final h lives in hA after layer 2)
  const long long poolN = (long long)NB * HDIM + NB;
  zero_f32<<<nblk(poolN, T), T, 0, stream>>>(pooled, poolN);  // pooled and cnt are contiguous
  pool_scatter<<<nblk((long long)NT[0] * 32, T), T, 0, stream>>>(
      hA[0], (const int*)d_in[IX_BATCH], pooled, cnt, NT[0]);
  final_lin<<<nblk(NB, T), T, 0, stream>>>(pooled, cnt, (const float*)d_in[IX_OUTW],
                                           (const float*)d_in[IX_OUTB], (float*)d_out, NB);
}
